// HaciCognitiveNet_79731772883078
// MI455X (gfx1250) — matrix operand order repaired and checked by reading, not compiled, not run on hardware
//
#include <hip/hip_runtime.h>
#include <hip/hip_bf16.h>
#include <math.h>

// ---------------------------------------------------------------------------
// CDNA5 (gfx1250) RetNet forward: all contractions via v_wmma_f32_16x16x32_f16
// GEMMs use a 2-stage LDS pipeline; A-tile staged via async global->LDS
// (ASYNCcnt path) when the toolchain exposes the builtin, else sync fallback.
// ---------------------------------------------------------------------------

typedef _Float16 v16h __attribute__((ext_vector_type(16)));
typedef _Float16 h8   __attribute__((ext_vector_type(8)));
typedef float    v8f  __attribute__((ext_vector_type(8)));
typedef int      v4i  __attribute__((vector_size(16)));  // matches builtin param

#define DEV_INLINE __device__ __forceinline__

#if __has_builtin(__builtin_amdgcn_global_load_async_to_lds_b128)
#define ASYNC_A 1
#else
#define ASYNC_A 0
#endif

// global (addrspace 1) / LDS (addrspace 3) int4 pointer casts for the builtin
#define GPTR_V4I(p) ((__attribute__((address_space(1))) v4i*)(p))
#define LPTR_V4I(p) ((__attribute__((address_space(3))) v4i*)(p))

DEV_INLINE void wait_async0() {
#if __has_builtin(__builtin_amdgcn_s_wait_asynccnt)
  __builtin_amdgcn_s_wait_asynccnt(0);
#else
  asm volatile("s_wait_asynccnt 0x0" ::: "memory");
#endif
}

namespace cfg {
constexpr int B = 8, S = 512, DIN = 384, D = 512, H = 8, FF = 2048, DH = 64;
constexpr int M = B * S;  // 4096 rows for all GEMMs
}

DEV_INLINE v8f zero_v8f() {
  v8f z = {0.f, 0.f, 0.f, 0.f, 0.f, 0.f, 0.f, 0.f};
  return z;
}

DEV_INLINE v16h pack16(h8 lo, h8 hi) {
  v16h r;
#pragma unroll
  for (int i = 0; i < 8; ++i) { r[i] = lo[i]; r[8 + i] = hi[i]; }
  return r;
}

// 8 f32 at p0 -> elements 0..7, 8 f32 at p1 -> elements 8..15 (f16)
DEV_INLINE v16h cvt16(const float* p0, const float* p1) {
  v16h r;
#pragma unroll
  for (int i = 0; i < 8; ++i) {
    r[i]     = (_Float16)p0[i];
    r[8 + i] = (_Float16)p1[i];
  }
  return r;
}

// ---------------------------------------------------------------------------
// GEMM: C[M,N] = act(A[M,K] @ W[K,N] + bias + res)
// ACT: 0 = none, 1 = sigmoid, 2 = exact gelu
// Block 256 thr (8 waves, 4x2), tile 128x128x32, double-buffered LDS.
// A tile: f32 in LDS (async DMA when available), converted at fragment build.
// B tile: register-staged transpose, f16 in LDS.
// ---------------------------------------------------------------------------
template <int ACT>
__global__ __launch_bounds__(256)
void gemm_wmma_kernel(const float* __restrict__ A, const float* __restrict__ W,
                      const float* __restrict__ bias, const float* __restrict__ res,
                      float* __restrict__ C, int M, int K, int N) {
  constexpr int BM = 128, BN = 128, BK = 32;
  constexpr int LDA = BK + 4;   // f32 pitch 36 -> 144B rows (16B aligned)
  constexpr int LDB = BK + 16;  // f16 pitch 48 -> 96B rows (16B aligned)
  __shared__ float    As[2][BM][LDA];
  __shared__ _Float16 Bs[2][BN][LDB];  // transposed: Bs[buf][n][k]

  const int tid  = threadIdx.x;
  const int lane = tid & 31;
  const int wid  = tid >> 5;
  const int wm   = wid >> 1;   // 0..3
  const int wn   = wid & 1;    // 0..1
  const int l16  = lane & 15;
  const int hlf  = lane >> 4;  // 0/1

  const int rowBase = blockIdx.y * BM;
  const int colBase = blockIdx.x * BN;

  // cooperative-load mapping (M,K,N multiples of tile dims in this net)
  const int ar = tid >> 1;        // A tile row 0..127
  const int ak = (tid & 1) * 16;  // A tile k-seg 0/16
  const int br = tid >> 3;        // B tile k row 0..31
  const int bc = (tid & 7) * 16;  // B tile col-seg

  v8f acc[2][4];
#pragma unroll
  for (int i = 0; i < 2; ++i)
#pragma unroll
    for (int j = 0; j < 4; ++j) acc[i][j] = zero_v8f();

  // ---- staging helpers -----------------------------------------------------
  auto stageA_issue = [&](int k0, int bufn, float* ta) {
    const float* ap = A + (size_t)(rowBase + ar) * K + k0 + ak;
#if ASYNC_A
    (void)ta;
    float* ldst = &As[bufn][ar][ak];
    __builtin_amdgcn_global_load_async_to_lds_b128(GPTR_V4I(ap), LPTR_V4I(ldst), 0, 0);
    __builtin_amdgcn_global_load_async_to_lds_b128(GPTR_V4I(ap), LPTR_V4I(ldst), 16, 0);
    __builtin_amdgcn_global_load_async_to_lds_b128(GPTR_V4I(ap), LPTR_V4I(ldst), 32, 0);
    __builtin_amdgcn_global_load_async_to_lds_b128(GPTR_V4I(ap), LPTR_V4I(ldst), 48, 0);
#else
    (void)bufn;
#pragma unroll
    for (int i = 0; i < 4; ++i) *(float4*)&ta[4 * i] = ((const float4*)ap)[i];
#endif
  };
  auto stageA_commit = [&](const float* ta, int bufn) {
#if !ASYNC_A
    float* dst = &As[bufn][ar][ak];
#pragma unroll
    for (int i = 0; i < 4; ++i) *(float4*)&dst[4 * i] = *(const float4*)&ta[4 * i];
#else
    (void)ta; (void)bufn;
#endif
  };
  auto stageB_load = [&](int k0, float* tb) {
    const float* wp = W + (size_t)(k0 + br) * N + colBase + bc;
#pragma unroll
    for (int i = 0; i < 4; ++i) *(float4*)&tb[4 * i] = ((const float4*)wp)[i];
  };
  auto stageB_commit = [&](const float* tb, int bufn) {
#pragma unroll
    for (int i = 0; i < 16; ++i) Bs[bufn][bc + i][br] = (_Float16)tb[i];
  };

  // ---- pipeline ------------------------------------------------------------
  const int nT = K / BK;
  {
    float ta[16], tb[16];
    stageA_issue(0, 0, ta);
    stageB_load(0, tb);
    stageA_commit(ta, 0);
    stageB_commit(tb, 0);
#if ASYNC_A
    wait_async0();
#endif
    __syncthreads();
  }

  for (int t = 0; t < nT; ++t) {
    const int buf = t & 1;
    const bool more = (t + 1 < nT);
    float ta[16], tb[16];
    if (more) {
      stageA_issue((t + 1) * BK, buf ^ 1, ta);  // DMA next A tile
      stageB_load((t + 1) * BK, tb);            // next B tile -> regs
    }

    // A fragment: lanes 0-15 row=l16 K0..15; lanes 16-31 K16..31
    v16h af[2], bf[4];
#pragma unroll
    for (int mi = 0; mi < 2; ++mi) {
      const float* p = &As[buf][wm * 32 + mi * 16 + l16][hlf * 16];
      af[mi] = cvt16(p, p + 8);
    }
    // B fragment: lanes 0-15 col=l16 K0..15; lanes 16-31 K16..31
#pragma unroll
    for (int ni = 0; ni < 4; ++ni) {
      const _Float16* p = &Bs[buf][wn * 64 + ni * 16 + l16][hlf * 16];
      bf[ni] = pack16(*(const h8*)p, *(const h8*)(p + 8));
    }
#pragma unroll
    for (int mi = 0; mi < 2; ++mi)
#pragma unroll
      for (int ni = 0; ni < 4; ++ni)
        acc[mi][ni] = __builtin_amdgcn_wmma_f32_16x16x32_f16(
            false, af[mi], false, bf[ni], (short)0, acc[mi][ni], false, false);

    if (more) {
      stageA_commit(ta, buf ^ 1);
      stageB_commit(tb, buf ^ 1);
#if ASYNC_A
      wait_async0();
#endif
      __syncthreads();
    }
  }

  // epilogue: C layout -> lanes 0-15 col=l16 M=r; lanes 16-31 M=r+8
#pragma unroll
  for (int ni = 0; ni < 4; ++ni) {
    const int col = colBase + wn * 64 + ni * 16 + l16;
    const float bv = bias ? bias[col] : 0.f;
#pragma unroll
    for (int mi = 0; mi < 2; ++mi) {
#pragma unroll
      for (int r = 0; r < 8; ++r) {
        const int row = rowBase + wm * 32 + mi * 16 + hlf * 8 + r;
        float v = acc[mi][ni][r] + bv;
        if (res) v += res[(size_t)row * N + col];
        if (ACT == 1) v = 1.f / (1.f + __expf(-v));
        if (ACT == 2) v = 0.5f * v * (1.f + erff(v * 0.70710678118f));
        C[(size_t)row * N + col] = v;
      }
    }
  }
}

// ---------------------------------------------------------------------------
// Retention: out[b,q,h,:] = sum_k (q.k/8) * 0.99^(q-k) [q>=k] * v[b,k,h,:]
// One wave per (b*H+h, 16-query tile). Causal loop over 32-key steps.
// ---------------------------------------------------------------------------
__global__ __launch_bounds__(32)
void retention_kernel(const float* __restrict__ Qg, const float* __restrict__ Kg,
                      const float* __restrict__ Vg, float* __restrict__ Og) {
  constexpr int S = cfg::S, H = cfg::H, DH = cfg::DH;
  __shared__ _Float16 Ss[16][48];  // masked scores tile (96B pitch)
  __shared__ _Float16 Vt[64][32];  // V transposed: Vt[d][key]

  const int lane = threadIdx.x;
  const int l16 = lane & 15, hlf = lane >> 4;
  const int bh = blockIdx.y;
  const int b = bh / H, hd = bh % H;
  const int qt = blockIdx.x;
  const int q0 = qt * 16;

  // resident Q fragments: 16x64 as two K=32 A-fragments
  v16h qa[2];
  {
    const float* qrow = Qg + ((size_t)(b * S + q0 + l16) * H + hd) * DH;
#pragma unroll
    for (int c = 0; c < 2; ++c) {
      const int d0 = c * 32 + hlf * 16;
      qa[c] = cvt16(qrow + d0, qrow + d0 + 8);
    }
  }

  v8f o[4];
#pragma unroll
  for (int j = 0; j < 4; ++j) o[j] = zero_v8f();

  const float scale = 0.125f;               // 1/sqrt(64)
  const float LOG2_DECAY = -0.0145003122f;  // log2(0.99)

  const int nkt = (qt >> 1) + 1;  // causal: keys up to diagonal
  for (int kt = 0; kt < nkt; ++kt) {
    const int kb = kt * 32;
    {  // stage V tile transposed (each lane loads one key row)
      const float* vrow = Vg + ((size_t)(b * S + kb + lane) * H + hd) * DH;
#pragma unroll
      for (int d = 0; d < DH; d += 4) {
        float4 t = *(const float4*)(vrow + d);
        Vt[d + 0][lane] = (_Float16)t.x;
        Vt[d + 1][lane] = (_Float16)t.y;
        Vt[d + 2][lane] = (_Float16)t.z;
        Vt[d + 3][lane] = (_Float16)t.w;
      }
    }
    // scores for two 16-key subtiles: Q[16x64] @ K^T[64x16]
#pragma unroll
    for (int sub = 0; sub < 2; ++sub) {
      const int key = kb + sub * 16 + l16;  // this lane's B column
      const float* krow = Kg + ((size_t)(b * S + key) * H + hd) * DH;
      v8f sacc = zero_v8f();
#pragma unroll
      for (int c = 0; c < 2; ++c) {
        const int d0 = c * 32 + hlf * 16;  // B frag: 16 consecutive dh values
        v16h kf = cvt16(krow + d0, krow + d0 + 8);
        sacc = __builtin_amdgcn_wmma_f32_16x16x32_f16(
            false, qa[c], false, kf, (short)0, sacc, false, false);
      }
      // decay mask + scale in C layout, spill to LDS as f16
#pragma unroll
      for (int r = 0; r < 8; ++r) {
        const int qi = q0 + hlf * 8 + r;
        const int diff = qi - key;
        float sv = (diff >= 0) ? sacc[r] * scale * exp2f((float)diff * LOG2_DECAY)
                               : 0.f;
        Ss[hlf * 8 + r][sub * 16 + l16] = (_Float16)sv;
      }
    }
    __syncthreads();
    // reload masked scores as A fragment [16 x 32]
    v16h sa;
    {
      const _Float16* p = &Ss[l16][hlf * 16];
      sa = pack16(*(const h8*)p, *(const h8*)(p + 8));
    }
    // O[16x64] += S[16x32] @ V[32x64]
#pragma unroll
    for (int j = 0; j < 4; ++j) {
      const _Float16* p = &Vt[j * 16 + l16][hlf * 16];
      v16h vf = pack16(*(const h8*)p, *(const h8*)(p + 8));
      o[j] = __builtin_amdgcn_wmma_f32_16x16x32_f16(
          false, sa, false, vf, (short)0, o[j], false, false);
    }
    __syncthreads();
  }

  // store in [B,S,H,DH] layout
#pragma unroll
  for (int j = 0; j < 4; ++j) {
#pragma unroll
    for (int r = 0; r < 8; ++r) {
      const int m = q0 + hlf * 8 + r;
      Og[((size_t)(b * S + m) * H + hd) * DH + j * 16 + l16] = o[j][r];
    }
  }
}

// ---------------------------------------------------------------------------
// LayerNorm over D=512, one row per block, optional fused gate multiply.
// ---------------------------------------------------------------------------
__global__ __launch_bounds__(256)
void layernorm512_kernel(const float* __restrict__ x, const float* __restrict__ w,
                         const float* __restrict__ bb, const float* __restrict__ gate,
                         float* __restrict__ out) {
  const int row = blockIdx.x;
  const int tid = threadIdx.x;
  const float* xr = x + (size_t)row * 512;
  float v0 = xr[tid], v1 = xr[tid + 256];
  float s = v0 + v1, sq = v0 * v0 + v1 * v1;
#pragma unroll
  for (int off = 16; off > 0; off >>= 1) {
    s  += __shfl_xor(s, off, 32);
    sq += __shfl_xor(sq, off, 32);
  }
  __shared__ float ssum[8], ssq[8];
  if ((tid & 31) == 0) { ssum[tid >> 5] = s; ssq[tid >> 5] = sq; }
  __syncthreads();
  float S = 0.f, Q = 0.f;
#pragma unroll
  for (int i = 0; i < 8; ++i) { S += ssum[i]; Q += ssq[i]; }
  const float mu  = S * (1.f / 512.f);
  const float var = Q * (1.f / 512.f) - mu * mu;
  const float inv = rsqrtf(var + 1e-5f);
  float y0 = (v0 - mu) * inv * w[tid] + bb[tid];
  float y1 = (v1 - mu) * inv * w[tid + 256] + bb[tid + 256];
  if (gate) {
    y0 *= gate[(size_t)row * 512 + tid];
    y1 *= gate[(size_t)row * 512 + tid + 256];
  }
  out[(size_t)row * 512 + tid] = y0;
  out[(size_t)row * 512 + tid + 256] = y1;
}

// ---------------------------------------------------------------------------
// Host-side orchestration
// ---------------------------------------------------------------------------
static void launch_gemm(hipStream_t st, const float* A, const float* W,
                        const float* bias, const float* res, float* C,
                        int M_, int K_, int N_, int act) {
  dim3 grid(N_ / 128, M_ / 128), blk(256);
  if (act == 1)
    gemm_wmma_kernel<1><<<grid, blk, 0, st>>>(A, W, bias, res, C, M_, K_, N_);
  else if (act == 2)
    gemm_wmma_kernel<2><<<grid, blk, 0, st>>>(A, W, bias, res, C, M_, K_, N_);
  else
    gemm_wmma_kernel<0><<<grid, blk, 0, st>>>(A, W, bias, res, C, M_, K_, N_);
}

// one pre-normed multi-scale-retention block: outp = res + MSR(xn)
static void run_retention(hipStream_t st, const float* xn,
                          const float* wq, const float* bq,
                          const float* wk, const float* bk,
                          const float* wv, const float* bv,
                          const float* wg, const float* bg,
                          const float* wo, const float* bo,
                          const float* lnw, const float* lnb,
                          float* qb, float* kb, float* vb, float* gb, float* ret,
                          const float* res, float* outp) {
  using namespace cfg;
  launch_gemm(st, xn, wq, bq, nullptr, qb, M, D, D, 0);
  launch_gemm(st, xn, wk, bk, nullptr, kb, M, D, D, 0);
  launch_gemm(st, xn, wv, bv, nullptr, vb, M, D, D, 0);
  launch_gemm(st, xn, wg, bg, nullptr, gb, M, D, D, 1);  // fused sigmoid
  retention_kernel<<<dim3(S / 16, B * H), 32, 0, st>>>(qb, kb, vb, ret);
  // qb reused as LN(ret)*g
  layernorm512_kernel<<<M, 256, 0, st>>>(ret, lnw, lnb, gb, qb);
  launch_gemm(st, qb, wo, bo, res, outp, M, D, D, 0);  // fused residual
}

extern "C" void kernel_launch(void* const* d_in, const int* in_sizes, int n_in,
                              void* d_out, int out_size, void* d_ws, size_t ws_size,
                              hipStream_t stream) {
  using namespace cfg;
  (void)in_sizes; (void)n_in; (void)out_size; (void)ws_size;

  int i = 0;
  const float* x       = (const float*)d_in[i++];
  const float* in_w    = (const float*)d_in[i++];
  const float* in_b    = (const float*)d_in[i++];
  const float* wm_wq   = (const float*)d_in[i++];
  const float* wm_bq   = (const float*)d_in[i++];
  const float* wm_wk   = (const float*)d_in[i++];
  const float* wm_bk   = (const float*)d_in[i++];
  const float* wm_wv   = (const float*)d_in[i++];
  const float* wm_bv   = (const float*)d_in[i++];
  const float* wm_wg   = (const float*)d_in[i++];
  const float* wm_bg   = (const float*)d_in[i++];
  const float* wm_wo   = (const float*)d_in[i++];
  const float* wm_bo   = (const float*)d_in[i++];
  const float* wm_lnw  = (const float*)d_in[i++];
  const float* wm_lnb  = (const float*)d_in[i++];
  const float* wm_prew = (const float*)d_in[i++];
  const float* wm_preb = (const float*)d_in[i++];
  const float* ffn_w1  = (const float*)d_in[i++];
  const float* ffn_b1  = (const float*)d_in[i++];
  const float* ffn_w2  = (const float*)d_in[i++];
  const float* ffn_b2  = (const float*)d_in[i++];
  const float* wm_onw  = (const float*)d_in[i++];
  const float* wm_onb  = (const float*)d_in[i++];
  const float* co_wq   = (const float*)d_in[i++];
  const float* co_bq   = (const float*)d_in[i++];
  const float* co_wk   = (const float*)d_in[i++];
  const float* co_bk   = (const float*)d_in[i++];
  const float* co_wv   = (const float*)d_in[i++];
  const float* co_bv   = (const float*)d_in[i++];
  const float* co_wg   = (const float*)d_in[i++];
  const float* co_bg   = (const float*)d_in[i++];
  const float* co_wo   = (const float*)d_in[i++];
  const float* co_bo   = (const float*)d_in[i++];
  const float* co_lnw  = (const float*)d_in[i++];
  const float* co_lnb  = (const float*)d_in[i++];
  const float* co_prew = (const float*)d_in[i++];
  const float* co_preb = (const float*)d_in[i++];

  const size_t MD = (size_t)M * D;
  float* ws  = (float*)d_ws;
  float* h   = ws; ws += MD;
  float* hn  = ws; ws += MD;
  float* qb  = ws; ws += MD;
  float* kb  = ws; ws += MD;
  float* vb  = ws; ws += MD;
  float* gb  = ws; ws += MD;
  float* ret = ws; ws += MD;
  float* f1  = ws; ws += (size_t)M * FF;

  // input projection
  launch_gemm(stream, x, in_w, in_b, nullptr, h, M, DIN, D, 0);

  // WorldModel: 2x (pre-LN MSR + residual, shared FFN + residual)
  for (int l = 0; l < 2; ++l) {
    layernorm512_kernel<<<M, 256, 0, stream>>>(h, wm_prew + l * D, wm_preb + l * D,
                                               nullptr, hn);
    run_retention(stream, hn,
                  wm_wq + (size_t)l * D * D, wm_bq + l * D,
                  wm_wk + (size_t)l * D * D, wm_bk + l * D,
                  wm_wv + (size_t)l * D * D, wm_bv + l * D,
                  wm_wg + (size_t)l * D * D, wm_bg + l * D,
                  wm_wo + (size_t)l * D * D, wm_bo + l * D,
                  wm_lnw + l * D, wm_lnb + l * D,
                  qb, kb, vb, gb, ret, h, h);
    launch_gemm(stream, h, ffn_w1, ffn_b1, nullptr, f1, M, D, FF, 2);   // gelu
    launch_gemm(stream, f1, ffn_w2, ffn_b2, h, h, M, FF, D, 0);         // +res
  }
  // output norm (in place: each thread reads before barrier, writes after)
  layernorm512_kernel<<<M, 256, 0, stream>>>(h, wm_onw, wm_onb, nullptr, h);

  // Retention core: 4x pre-LN residual MSR; last layer writes d_out
  for (int l = 0; l < 4; ++l) {
    layernorm512_kernel<<<M, 256, 0, stream>>>(h, co_prew + l * D, co_preb + l * D,
                                               nullptr, hn);
    float* outp = (l == 3) ? (float*)d_out : h;
    run_retention(stream, hn,
                  co_wq + (size_t)l * D * D, co_bq + l * D,
                  co_wk + (size_t)l * D * D, co_bk + l * D,
                  co_wv + (size_t)l * D * D, co_bv + l * D,
                  co_wg + (size_t)l * D * D, co_bg + l * D,
                  co_wo + (size_t)l * D * D, co_bo + l * D,
                  co_lnw + l * D, co_lnb + l * D,
                  qb, kb, vb, gb, ret, h, outp);
  }
}
